// Mamba2_48971217109702
// MI455X (gfx1250) — compile-verified, matrix operand order checked by reading, not run on hardware
//
#include <hip/hip_runtime.h>
#include <hip/hip_bf16.h>

#define B_ 4
#define L_ 2048
#define D_MODEL 1024
#define D_STATE 128
#define D_CONV 4
#define HEADDIM 64
#define CHUNK 256
#define D_INNER 2048
#define NHEADS 32
#define CONV_DIM 2304
#define D_IN_PROJ 4384
#define NCHUNK 8
#define BL (B_*L_)

typedef __bf16 bf16_t;
typedef __attribute__((ext_vector_type(16))) __bf16 v16bf;
typedef __attribute__((ext_vector_type(8)))  float   v8f;

#define LP 40  // LDS row stride in bf16 elements (80B: avoids bank-conflict periodicity)

// Load a 16x32 bf16 WMMA fragment from an LDS tile stored (rows x K) row-major.
// Works for both A (M x K) and B (N x K, i.e. W of C=A*W^T) operands: identical lane layout.
__device__ __forceinline__ v16bf frag_ld(const bf16_t* base, int row, int stride) {
  const int lane = threadIdx.x & 31;
  const bf16_t* p = base + (row + (lane & 15)) * stride + ((lane >> 4) << 3);
  union { v16bf v; uint4 q[2]; } u;
  u.q[0] = *(const uint4*)(p);
  u.q[1] = *(const uint4*)(p + 16);
  return u.v;
}

__device__ __forceinline__ unsigned pk2(float a, float b) {
  union { bf16_t h[2]; unsigned u; } x;
  x.h[0] = (bf16_t)a; x.h[1] = (bf16_t)b;
  return x.u;
}

__device__ __forceinline__ v8f wmma_bf16(v16bf a, v16bf b, v8f c) {
  return __builtin_amdgcn_wmma_f32_16x16x32_bf16(false, a, false, b, (short)0, c, false, false);
}

// Async global->LDS 16B copy (CDNA5 VGLOBAL op 98, ASYNCcnt-tracked).
// ldsoff = wave-relative LDS byte offset (low 32 bits of a flat LDS pointer).
__device__ __forceinline__ void async_ld_b128(unsigned ldsoff, const void* gaddr) {
  asm volatile("global_load_async_to_lds_b128 %0, %1, off" :: "v"(ldsoff), "v"(gaddr) : "memory");
}
__device__ __forceinline__ void wait_async0() {
  asm volatile("s_wait_asynccnt 0x0" ::: "memory");
}

// fp32 -> bf16 bulk conversion (weights, done once per launch).
__global__ __launch_bounds__(256) void cvt_f32_bf16(const float* __restrict__ s,
                                                    bf16_t* __restrict__ d, long n) {
  long i = ((long)blockIdx.x * 256 + threadIdx.x) * 4;
  if (i >= n) return;
  float4 f = *(const float4*)(s + i);
  uint2 o; o.x = pk2(f.x, f.y); o.y = pk2(f.z, f.w);
  *(uint2*)(d + i) = o;
}

// ---------------------------------------------------------------------------
// NT GEMM with bf16 weights: C[M,N] = A[M,K](f32) * W[N,K](bf16)^T.
// 128x128 block tile, 8 waves (4Mx2N), 2x4 WMMA tiles/wave.
// Double-buffered LDS, one barrier per K-step; W tile staged with
// global_load_async_to_lds_b128, A tile converted f32->bf16 through regs.
// M,K multiples of 128/32; N guarded (source rows clamped, stores masked).
// ---------------------------------------------------------------------------
__global__ __launch_bounds__(256) void gemm_nt_wb(
    const float* __restrict__ A, const bf16_t* __restrict__ W, float* __restrict__ C,
    int M, int N, int K, int lda, int ldw, int ldc)
{
  __shared__ bf16_t As[2][128 * LP];
  __shared__ bf16_t Ws[2][128 * LP];
  const int m0 = blockIdx.y * 128, n0 = blockIdx.x * 128;
  const int t = threadIdx.x, lane = t & 31, wid = t >> 5;
  const int wm = wid & 3, wn = wid >> 2;
  v8f acc[2][4] = {};
  const int srow = t >> 1, sc0 = (t & 1) * 16;
  float4 areg[4];

  auto load_A = [&](int k0) {
    const float* pa = A + (long)(m0 + srow) * lda + k0 + sc0;
#pragma unroll
    for (int q = 0; q < 4; ++q) areg[q] = ((const float4*)pa)[q];
  };
  auto store_A = [&](int buf) {
    unsigned* da = (unsigned*)&As[buf][srow * LP + sc0];
#pragma unroll
    for (int q = 0; q < 4; ++q) {
      da[2*q+0] = pk2(areg[q].x, areg[q].y);
      da[2*q+1] = pk2(areg[q].z, areg[q].w);
    }
  };
  auto stage_W = [&](int k0, int buf) {
#pragma unroll
    for (int e = 0; e < 2; ++e) {
      int q = t * 2 + e;            // 512 16B segments: 128 rows x 4 segs
      int row = q >> 2, seg = q & 3;
      int gr = n0 + row; if (gr > N - 1) gr = N - 1;   // clamp (results masked at store)
      const bf16_t* gp = W + (long)gr * ldw + k0 + seg * 8;
      unsigned lo = (unsigned)(size_t)&Ws[buf][row * LP + seg * 8];
      async_ld_b128(lo, gp);
    }
  };

  // prologue: stage tile 0
  load_A(0);
  stage_W(0, 0);
  store_A(0);
  wait_async0();
  __syncthreads();

  const int nsteps = K / 32;
  for (int s = 0; s < nsteps; ++s) {
    const int p = s & 1;
    const bool more = (s + 1 < nsteps);
    if (more) {           // buf p^1 is dead: its readers consumed it before last barrier
      load_A((s + 1) * 32);
      stage_W((s + 1) * 32, p ^ 1);
    }
    v16bf af[2], wf[4];
#pragma unroll
    for (int i = 0; i < 2; ++i) af[i] = frag_ld(&As[p][0], wm*32 + i*16, LP);
#pragma unroll
    for (int j = 0; j < 4; ++j) wf[j] = frag_ld(&Ws[p][0], wn*64 + j*16, LP);
#pragma unroll
    for (int i = 0; i < 2; ++i)
#pragma unroll
      for (int j = 0; j < 4; ++j) acc[i][j] = wmma_bf16(af[i], wf[j], acc[i][j]);
    if (more) {
      store_A(p ^ 1);
      wait_async0();
    }
    __syncthreads();
  }

  const int rb = (lane >> 4) << 3, cc = lane & 15;
#pragma unroll
  for (int i = 0; i < 2; ++i) {
    int rowb = m0 + wm*32 + i*16 + rb;
#pragma unroll
    for (int j = 0; j < 4; ++j) {
      int col = n0 + wn*64 + j*16 + cc;
      if (col < N) {
#pragma unroll
        for (int r = 0; r < 8; ++r)
          C[(long)(rowb + r) * ldc + col] = acc[i][j][r];
      }
    }
  }
}

// ---------------------------------------------------------------------------
// Generic batched NT GEMM (fp32 A and W), used for the small CB batch.
// ---------------------------------------------------------------------------
__global__ __launch_bounds__(256) void gemm_nt(
    const float* __restrict__ A, const float* __restrict__ W, float* __restrict__ C,
    int M, int N, int K, int lda, int ldw, int ldc,
    long sA, long sW, long sC)
{
  __shared__ bf16_t As[128 * LP];
  __shared__ bf16_t Ws[128 * LP];
  A += sA * blockIdx.z; W += sW * blockIdx.z; C += sC * blockIdx.z;
  const int m0 = blockIdx.y * 128, n0 = blockIdx.x * 128;
  const int t = threadIdx.x, lane = t & 31, wid = t >> 5;
  const int wm = wid & 3, wn = wid >> 2;
  v8f acc[2][4] = {};
  const int srow = t >> 1, sc0 = (t & 1) * 16;
  for (int k0 = 0; k0 < K; k0 += 32) {
    {
      const float* pa = A + (long)(m0 + srow) * lda + k0 + sc0;
      unsigned* da = (unsigned*)&As[srow * LP + sc0];
#pragma unroll
      for (int q = 0; q < 4; ++q) {
        float4 f = ((const float4*)pa)[q];
        da[2*q+0] = pk2(f.x, f.y);
        da[2*q+1] = pk2(f.z, f.w);
      }
      unsigned* dw = (unsigned*)&Ws[srow * LP + sc0];
      if (n0 + srow < N) {
        const float* pw = W + (long)(n0 + srow) * ldw + k0 + sc0;
#pragma unroll
        for (int q = 0; q < 4; ++q) {
          float4 f = ((const float4*)pw)[q];
          dw[2*q+0] = pk2(f.x, f.y);
          dw[2*q+1] = pk2(f.z, f.w);
        }
      } else {
#pragma unroll
        for (int q = 0; q < 8; ++q) dw[q] = 0u;
      }
    }
    __syncthreads();
    v16bf af[2], wf[4];
#pragma unroll
    for (int i = 0; i < 2; ++i) af[i] = frag_ld(As, wm*32 + i*16, LP);
#pragma unroll
    for (int j = 0; j < 4; ++j) wf[j] = frag_ld(Ws, wn*64 + j*16, LP);
#pragma unroll
    for (int i = 0; i < 2; ++i)
#pragma unroll
      for (int j = 0; j < 4; ++j) acc[i][j] = wmma_bf16(af[i], wf[j], acc[i][j]);
    __syncthreads();
  }
  const int rb = (lane >> 4) << 3;
  const int cc = lane & 15;
#pragma unroll
  for (int i = 0; i < 2; ++i) {
    int rowb = m0 + wm*32 + i*16 + rb;
#pragma unroll
    for (int j = 0; j < 4; ++j) {
      int col = n0 + wn*64 + j*16 + cc;
      if (col < N) {
#pragma unroll
        for (int r = 0; r < 8; ++r)
          C[(long)(rowb + r) * ldc + col] = acc[i][j][r];
      }
    }
  }
}

// ---------------------------------------------------------------------------
// Depthwise causal conv (width 4) + bias + SiLU over xBC slice of zxbcdt.
// ---------------------------------------------------------------------------
__global__ __launch_bounds__(256) void conv_silu(
    const float* __restrict__ zx, const float* __restrict__ cw,
    const float* __restrict__ cb, float* __restrict__ xact)
{
  long idx = (long)blockIdx.x * 256 + threadIdx.x;
  if (idx >= (long)BL * CONV_DIM) return;
  int cch = (int)(idx % CONV_DIM);
  long row = idx / CONV_DIM;
  int lb = (int)(row % L_);
  float a = cb[cch];
#pragma unroll
  for (int k = 0; k < D_CONV; ++k) {
    int ll = lb - (D_CONV - 1) + k;
    if (ll >= 0)
      a += zx[(row - lb + ll) * D_IN_PROJ + D_INNER + cch] * cw[cch * D_CONV + k];
  }
  xact[row * CONV_DIM + cch] = a / (1.f + __expf(-a));
}

__global__ void dt_softplus(const float* __restrict__ zx, const float* __restrict__ dtb,
                            float* __restrict__ dtv)
{
  int idx = blockIdx.x * 256 + threadIdx.x;
  if (idx >= BL * NHEADS) return;
  int h = idx % NHEADS;
  long row = idx / NHEADS;
  float x = zx[row * D_IN_PROJ + (D_INNER + CONV_DIM) + h] + dtb[h];
  dtv[idx] = (x > 20.f) ? x : log1pf(__expf(x));
}

__global__ void dA_cumsum(const float* __restrict__ dtv, const float* __restrict__ Alog,
                          float* __restrict__ dacs, float* __restrict__ cdec)
{
  int idx = blockIdx.x * 256 + threadIdx.x;
  if (idx >= B_ * NCHUNK * NHEADS) return;
  int h = idx % NHEADS;
  int c = (idx / NHEADS) % NCHUNK;
  int b = idx / (NHEADS * NCHUNK);
  float Ah = -__expf(Alog[h]);
  long lbase = (long)b * L_ + (long)c * CHUNK;
  long dbase = (long)(b * NCHUNK + c) * CHUNK;
  float cs = 0.f;
  for (int i = 0; i < CHUNK; ++i) {
    cs += dtv[(lbase + i) * NHEADS + h] * Ah;
    dacs[(dbase + i) * NHEADS + h] = cs;
  }
  cdec[(long)(b * NCHUNK + c) * NHEADS + h] = __expf(cs);
}

// ---------------------------------------------------------------------------
// Y_diag for one (b,c,h): Y = (L_h ⊙ CB)(256x256) @ (dt*x)(256x64), + D*x.
// ---------------------------------------------------------------------------
__global__ __launch_bounds__(256) void ssd_diag(
    const float* __restrict__ xact, const float* __restrict__ dtv,
    const float* __restrict__ dacs, const float* __restrict__ CB,
    const float* __restrict__ Dp, float* __restrict__ Y)
{
  __shared__ bf16_t Ss[256 * LP];
  __shared__ bf16_t Xs[64 * LP];
  __shared__ float dj[32], tj[32];
  const int h = blockIdx.x, c = blockIdx.y, b = blockIdx.z;
  const int t = threadIdx.x, lane = t & 31, w = t >> 5;
  const long lbase = (long)b * L_ + (long)c * CHUNK;
  const long dbase = (long)(b * NCHUNK + c) * CHUNK;
  const float* cbp = CB + (long)(b * NCHUNK + c) * (CHUNK * CHUNK);
  const float dacs_i = dacs[(dbase + t) * NHEADS + h];
  v8f acc[2][4] = {};
  for (int j0 = 0; j0 < CHUNK; j0 += 32) {
    if (t < 32) {
      int j = j0 + t;
      dj[t] = dacs[(dbase + j) * NHEADS + h];
      tj[t] = dtv[(lbase + j) * NHEADS + h];
    }
    __syncthreads();
    {  // S row i = t
      const int i = t;
      unsigned* ds = (unsigned*)&Ss[i * LP];
      const float* cbr = cbp + (long)i * CHUNK + j0;
#pragma unroll
      for (int q = 0; q < 16; ++q) {
        int ja = 2 * q, jb = 2 * q + 1;
        float va = (j0 + ja <= i) ? cbr[ja] * __expf(dacs_i - dj[ja]) : 0.f;
        float vb = (j0 + jb <= i) ? cbr[jb] * __expf(dacs_i - dj[jb]) : 0.f;
        ds[q] = pk2(va, vb);
      }
    }
    {  // (dt*x)^T tile: rows p, cols j
      const int p = t >> 2, jj0 = (t & 3) * 8;
      unsigned* dx = (unsigned*)&Xs[p * LP + jj0];
#pragma unroll
      for (int q = 0; q < 4; ++q) {
        int ja = jj0 + 2 * q, jb = ja + 1;
        float va = tj[ja] * xact[(lbase + j0 + ja) * CONV_DIM + h * HEADDIM + p];
        float vb = tj[jb] * xact[(lbase + j0 + jb) * CONV_DIM + h * HEADDIM + p];
        dx[q] = pk2(va, vb);
      }
    }
    __syncthreads();
    if (j0 <= w * 32 + 31) {  // causal: skip strictly-upper K tiles
      v16bf af[2], xf[4];
#pragma unroll
      for (int i = 0; i < 2; ++i) af[i] = frag_ld(Ss, w * 32 + i * 16, LP);
#pragma unroll
      for (int j = 0; j < 4; ++j) xf[j] = frag_ld(Xs, j * 16, LP);
#pragma unroll
      for (int i = 0; i < 2; ++i)
#pragma unroll
        for (int j = 0; j < 4; ++j) acc[i][j] = wmma_bf16(af[i], xf[j], acc[i][j]);
    }
    __syncthreads();
  }
  const int rb = (lane >> 4) << 3, cc = lane & 15;
  const float Dh = Dp[h];
#pragma unroll
  for (int i = 0; i < 2; ++i) {
    int rowb = w * 32 + i * 16 + rb;
#pragma unroll
    for (int j = 0; j < 4; ++j) {
      int p = j * 16 + cc;
#pragma unroll
      for (int r = 0; r < 8; ++r) {
        long l = lbase + rowb + r;
        float xv = xact[l * CONV_DIM + h * HEADDIM + p];
        Y[l * D_INNER + h * HEADDIM + p] = acc[i][j][r] + Dh * xv;
      }
    }
  }
}

// ---------------------------------------------------------------------------
// Chunk states for one (b,c,h): G[n,p] = sum_j B[j,n] * decay[j]*dt[j]*x[j,p].
// ---------------------------------------------------------------------------
__global__ __launch_bounds__(256) void ssd_states(
    const float* __restrict__ xact, const float* __restrict__ dtv,
    const float* __restrict__ dacs, float* __restrict__ states)
{
  __shared__ bf16_t Bs[128 * LP];
  __shared__ bf16_t Xs[64 * LP];
  __shared__ float sc[32];
  const int h = blockIdx.x, c = blockIdx.y, b = blockIdx.z;
  const int t = threadIdx.x, lane = t & 31, w = t >> 5;
  const long lbase = (long)b * L_ + (long)c * CHUNK;
  const long dbase = (long)(b * NCHUNK + c) * CHUNK;
  const float last = dacs[(dbase + CHUNK - 1) * NHEADS + h];
  v8f acc[4] = {};
  for (int j0 = 0; j0 < CHUNK; j0 += 32) {
    if (t < 32) {
      int j = j0 + t;
      float d = dacs[(dbase + j) * NHEADS + h];
      sc[t] = __expf(last - d) * dtv[(lbase + j) * NHEADS + h];
    }
    __syncthreads();
    {  // B^T tile: rows n, cols j
      const int n = t >> 1, jj0 = (t & 1) * 16;
      unsigned* db = (unsigned*)&Bs[n * LP + jj0];
#pragma unroll
      for (int q = 0; q < 8; ++q) {
        int ja = jj0 + 2 * q, jb = ja + 1;
        float va = xact[(lbase + j0 + ja) * CONV_DIM + D_INNER + n];
        float vb = xact[(lbase + j0 + jb) * CONV_DIM + D_INNER + n];
        db[q] = pk2(va, vb);
      }
    }
    {  // (decay*dt*x)^T tile: rows p, cols j
      const int p = t >> 2, jj0 = (t & 3) * 8;
      unsigned* dx = (unsigned*)&Xs[p * LP + jj0];
#pragma unroll
      for (int q = 0; q < 4; ++q) {
        int ja = jj0 + 2 * q, jb = ja + 1;
        float va = sc[ja] * xact[(lbase + j0 + ja) * CONV_DIM + h * HEADDIM + p];
        float vb = sc[jb] * xact[(lbase + j0 + jb) * CONV_DIM + h * HEADDIM + p];
        dx[q] = pk2(va, vb);
      }
    }
    __syncthreads();
    v16bf bfrag = frag_ld(Bs, w * 16, LP);
    v16bf xf[4];
#pragma unroll
    for (int j = 0; j < 4; ++j) xf[j] = frag_ld(Xs, j * 16, LP);
#pragma unroll
    for (int j = 0; j < 4; ++j) acc[j] = wmma_bf16(bfrag, xf[j], acc[j]);
    __syncthreads();
  }
  const int rb = (lane >> 4) << 3, cc = lane & 15;
  float* sp = states + ((long)(b * NCHUNK + c) * NHEADS + h) * (HEADDIM * D_STATE);
#pragma unroll
  for (int j = 0; j < 4; ++j) {
    int p = j * 16 + cc;
#pragma unroll
    for (int r = 0; r < 8; ++r) {
      int n = w * 16 + rb + r;
      sp[(long)p * D_STATE + n] = acc[j][r];  // states[b,c,h,p,n]
    }
  }
}

// Inter-chunk recurrence: prev[c] = carry; carry = carry*decay[c] + states[c].
__global__ __launch_bounds__(256) void chunk_scan(
    const float* __restrict__ states, const float* __restrict__ cdec,
    float* __restrict__ prev)
{
  const int bh = blockIdx.x, b = bh / NHEADS, h = bh % NHEADS;
  const int t = threadIdx.x;
  const int PN = HEADDIM * D_STATE;  // 8192
  float carry[PN / 256];
#pragma unroll
  for (int q = 0; q < PN / 256; ++q) carry[q] = 0.f;
  for (int c = 0; c < NCHUNK; ++c) {
    long base = ((long)(b * NCHUNK + c) * NHEADS + h) * PN;
    float d = cdec[(long)(b * NCHUNK + c) * NHEADS + h];
#pragma unroll
    for (int q = 0; q < PN / 256; ++q) {
      long e = base + t + q * 256;
      prev[e] = carry[q];
      carry[q] = carry[q] * d + states[e];
    }
  }
}

// ---------------------------------------------------------------------------
// Y_off for one (b,c,h): Y += exp(dA_cs[i]) * C(256x128) @ prev(64x128)^T.
// ---------------------------------------------------------------------------
__global__ __launch_bounds__(256) void ssd_off(
    const float* __restrict__ xact, const float* __restrict__ dacs,
    const float* __restrict__ prev, float* __restrict__ Y)
{
  __shared__ bf16_t Cs[256 * LP];
  __shared__ bf16_t Ps[64 * LP];
  const int h = blockIdx.x, c = blockIdx.y, b = blockIdx.z;
  const int t = threadIdx.x, lane = t & 31, w = t >> 5;
  const long lbase = (long)b * L_ + (long)c * CHUNK;
  const long dbase = (long)(b * NCHUNK + c) * CHUNK;
  const float* pv = prev + ((long)(b * NCHUNK + c) * NHEADS + h) * (HEADDIM * D_STATE);
  v8f acc[2][4] = {};
  for (int n0 = 0; n0 < D_STATE; n0 += 32) {
    {  // C tile: row i = t, 32 contiguous n
      const float* cr = xact + (lbase + t) * CONV_DIM + D_INNER + D_STATE + n0;
      unsigned* dc = (unsigned*)&Cs[t * LP];
#pragma unroll
      for (int q = 0; q < 16; ++q) dc[q] = pk2(cr[2 * q], cr[2 * q + 1]);
    }
    {  // prev tile: rows p, cols n (contiguous)
      const int p = t >> 2, nn0 = (t & 3) * 8;
      const float* pr = pv + (long)p * D_STATE + n0 + nn0;
      unsigned* dpp = (unsigned*)&Ps[p * LP + nn0];
#pragma unroll
      for (int q = 0; q < 4; ++q) dpp[q] = pk2(pr[2 * q], pr[2 * q + 1]);
    }
    __syncthreads();
    v16bf cf[2], pf[4];
#pragma unroll
    for (int i = 0; i < 2; ++i) cf[i] = frag_ld(Cs, w * 32 + i * 16, LP);
#pragma unroll
    for (int j = 0; j < 4; ++j) pf[j] = frag_ld(Ps, j * 16, LP);
#pragma unroll
    for (int i = 0; i < 2; ++i)
#pragma unroll
      for (int j = 0; j < 4; ++j) acc[i][j] = wmma_bf16(cf[i], pf[j], acc[i][j]);
    __syncthreads();
  }
  const int rb = (lane >> 4) << 3, cc = lane & 15;
#pragma unroll
  for (int i = 0; i < 2; ++i) {
    int rowb = w * 32 + i * 16 + rb;
#pragma unroll
    for (int r = 0; r < 8; ++r) {
      int row = rowb + r;
      float e = __expf(dacs[(dbase + row) * NHEADS + h]);
      long l = lbase + row;
#pragma unroll
      for (int j = 0; j < 4; ++j) {
        int p = j * 16 + cc;
        float* yp = &Y[l * D_INNER + h * HEADDIM + p];
        *yp = *yp + e * acc[i][j][r];
      }
    }
  }
}

// SiLU gate + RMSNorm over d_inner, one block per token row.
__global__ __launch_bounds__(256) void gate_rmsnorm(
    const float* __restrict__ Y, const float* __restrict__ zx,
    const float* __restrict__ nw, float* __restrict__ g)
{
  __shared__ float red[256];
  const long row = blockIdx.x;
  const int t = threadIdx.x;
  float loc[D_INNER / 256];
  float ss = 0.f;
#pragma unroll
  for (int q = 0; q < D_INNER / 256; ++q) {
    int d = t + q * 256;
    float z = zx[row * D_IN_PROJ + d];
    float v = Y[row * D_INNER + d] * (z / (1.f + __expf(-z)));
    loc[q] = v;
    ss += v * v;
  }
  red[t] = ss;
  __syncthreads();
  for (int s = 128; s > 0; s >>= 1) {
    if (t < s) red[t] += red[t + s];
    __syncthreads();
  }
  float rs = rsqrtf(red[0] / (float)D_INNER + 1e-5f);
#pragma unroll
  for (int q = 0; q < D_INNER / 256; ++q) {
    int d = t + q * 256;
    g[row * D_INNER + d] = loc[q] * rs * nw[d];
  }
}

extern "C" void kernel_launch(void* const* d_in, const int* in_sizes, int n_in,
                              void* d_out, int out_size, void* d_ws, size_t ws_size,
                              hipStream_t stream) {
  (void)in_sizes; (void)n_in; (void)out_size; (void)ws_size;
  const float* u       = (const float*)d_in[0];
  const float* W_in    = (const float*)d_in[1];
  const float* conv_w  = (const float*)d_in[2];
  const float* conv_b  = (const float*)d_in[3];
  const float* dt_bias = (const float*)d_in[4];
  const float* A_log   = (const float*)d_in[5];
  const float* D_par   = (const float*)d_in[6];
  const float* norm_w  = (const float*)d_in[7];
  const float* W_out   = (const float*)d_in[8];
  float* out = (float*)d_out;

  float* ws = (float*)d_ws;
  long off = 0;
  float* zx   = ws + off; off += (long)BL * D_IN_PROJ;                     // 143.7 MB
  float* xact = ws + off; off += (long)BL * CONV_DIM;                      // 75.5 MB
  float* dtv  = ws + off; off += (long)BL * NHEADS;
  float* dacs = ws + off; off += (long)B_ * NCHUNK * CHUNK * NHEADS;
  float* cdec = ws + off; off += (long)B_ * NCHUNK * NHEADS;
  float* CBb  = ws + off; off += (long)B_ * NCHUNK * CHUNK * CHUNK;       // 8 MB
  float* Yb   = ws + off; off += (long)BL * D_INNER;                      // 64 MB
  float* st   = ws + off; off += (long)B_ * NCHUNK * NHEADS * HEADDIM * D_STATE;
  float* pv   = ws + off; off += (long)B_ * NCHUNK * NHEADS * HEADDIM * D_STATE;
  bf16_t* Winb  = (bf16_t*)(ws + off); off += (long)D_IN_PROJ * D_MODEL / 2;
  bf16_t* Woutb = (bf16_t*)(ws + off); off += (long)D_MODEL * D_INNER / 2;
  float* g = xact;  // xact dead after ssd_off -> reuse for gated/normed activations

  // 0. one-time weight conversion to bf16
  long nwin = (long)D_IN_PROJ * D_MODEL, nwout = (long)D_MODEL * D_INNER;
  cvt_f32_bf16<<<(int)((nwin / 4 + 255) / 256), 256, 0, stream>>>(W_in, Winb, nwin);
  cvt_f32_bf16<<<(int)((nwout / 4 + 255) / 256), 256, 0, stream>>>(W_out, Woutb, nwout);

  // 1. in_proj: zxbcdt = u @ W_in^T   (async-LDS weight staging, double buffered)
  gemm_nt_wb<<<dim3((D_IN_PROJ + 127) / 128, BL / 128, 1), 256, 0, stream>>>(
      u, Winb, zx, BL, D_IN_PROJ, D_MODEL, D_MODEL, D_MODEL, D_IN_PROJ);
  // 2. depthwise conv + SiLU
  long nconv = (long)BL * CONV_DIM;
  conv_silu<<<(int)((nconv + 255) / 256), 256, 0, stream>>>(zx, conv_w, conv_b, xact);
  // 3. dt = softplus(dt_raw + bias)
  dt_softplus<<<(BL * NHEADS + 255) / 256, 256, 0, stream>>>(zx, dt_bias, dtv);
  // 4. per-chunk cumsum of dt*A, chunk decay
  dA_cumsum<<<(B_ * NCHUNK * NHEADS + 255) / 256, 256, 0, stream>>>(dtv, A_log, dacs, cdec);
  // 5. CB = C @ B^T per (b,chunk), batched
  gemm_nt<<<dim3(2, 2, B_ * NCHUNK), 256, 0, stream>>>(
      xact + (D_INNER + D_STATE), xact + D_INNER, CBb,
      CHUNK, CHUNK, D_STATE, CONV_DIM, CONV_DIM, CHUNK,
      (long)CHUNK * CONV_DIM, (long)CHUNK * CONV_DIM, (long)CHUNK * CHUNK);
  // 6. Y_diag (+ D*x)
  ssd_diag<<<dim3(NHEADS, NCHUNK, B_), 256, 0, stream>>>(xact, dtv, dacs, CBb, D_par, Yb);
  // 7. per-chunk end states
  ssd_states<<<dim3(NHEADS, NCHUNK, B_), 256, 0, stream>>>(xact, dtv, dacs, st);
  // 8. inter-chunk scan
  chunk_scan<<<B_ * NHEADS, 256, 0, stream>>>(st, cdec, pv);
  // 9. Y += Y_off
  ssd_off<<<dim3(NHEADS, NCHUNK, B_), 256, 0, stream>>>(xact, dacs, pv, Yb);
  // 10. gate + RMSNorm
  gate_rmsnorm<<<BL, 256, 0, stream>>>(Yb, zx, norm_w, g);
  // 11. out_proj: out = g @ W_out^T
  gemm_nt_wb<<<dim3(D_MODEL / 128, BL / 128, 1), 256, 0, stream>>>(
      g, Woutb, out, BL, D_MODEL, D_INNER, D_INNER, D_INNER, D_MODEL);
}
